// CodeAwareAttention_48996986912786
// MI455X (gfx1250) — matrix operand order repaired and checked by reading, not compiled, not run on hardware
//
#include <hip/hip_runtime.h>
#include <hip/hip_bf16.h>
#include <math.h>

// ---------------------------------------------------------------------------
// Types for CDNA5 WMMA (wave32): 16x16x32 bf16 -> f32
// ---------------------------------------------------------------------------
typedef __bf16 bf16;
typedef __attribute__((ext_vector_type(16))) __bf16 v16bf;
typedef __attribute__((ext_vector_type(8)))  __bf16 v8bf;
typedef __attribute__((ext_vector_type(8)))  float  v8f;

#define WMMA_BF16(a, b, c) \
  __builtin_amdgcn_wmma_f32_16x16x32_bf16(false, (a), false, (b), (short)0, (c), false, false)

// Problem constants (from reference)
#define BB 8
#define SS 768
#define DD 768
#define HH 12
#define DHD 64
#define MM (BB * SS)   // 6144 rows

// ---------------------------------------------------------------------------
// Fragment loaders per ISA §7.12.2 VGPR layouts (wave32)
// A (16x32, 16-bit): lane<16 -> row=lane, elems 0..7=K k0..k0+7, 8..15=k0+16..23
//                    lane>=16 -> row=lane-16, elems 0..7=k0+8..15, 8..15=k0+24..31
// ---------------------------------------------------------------------------
static __device__ inline v16bf frag_a_rowmajor(const bf16* base, int ld, int k0) {
  int lane = threadIdx.x & 31;
  int r    = lane & 15;
  int koff = (lane & 16) ? 8 : 0;
  const bf16* p = base + (size_t)r * ld + k0 + koff;
  v8bf lo = *(const v8bf*)(p);        // K = k0+koff   .. +7
  v8bf hi = *(const v8bf*)(p + 16);   // K = k0+koff+16.. +23
  v16bf f;
#pragma unroll
  for (int i = 0; i < 8; ++i) { f[i] = lo[i]; f[8 + i] = hi[i]; }
  return f;
}

// B (32x16, 16-bit) from N-major storage (each "column" n is a contiguous row
// of length K): lane<16 -> n=lane, elems 0..7 = K k0..k0+7, 8..15 = k0+16..23;
// lane>=16 -> n=lane-16, elems 0..7 = k0+8..15, 8..15 = k0+24..31.
static __device__ inline v16bf frag_b_nmajor(const bf16* base, int ld, int n0, int k0) {
  int lane = threadIdx.x & 31;
  int n    = lane & 15;
  int koff = (lane & 16) ? 8 : 0;
  const bf16* p = base + (size_t)(n0 + n) * ld + k0 + koff;
  v8bf lo = *(const v8bf*)(p);
  v8bf hi = *(const v8bf*)(p + 16);
  v16bf f;
#pragma unroll
  for (int i = 0; i < 8; ++i) { f[i] = lo[i]; f[8 + i] = hi[i]; }
  return f;
}

// A fragment sourced from an LDS fp32 buffer (probs), converting to bf16.
static __device__ inline v16bf frag_a_lds_f32(const float* s, int ld, int k0) {
  int lane = threadIdx.x & 31;
  int r    = lane & 15;
  int koff = (lane & 16) ? 8 : 0;
  const float* p = s + (size_t)r * ld + k0 + koff;
  v16bf f;
#pragma unroll
  for (int i = 0; i < 8; ++i) { f[i] = (bf16)p[i]; f[8 + i] = (bf16)p[i + 16]; }
  return f;
}

// ---------------------------------------------------------------------------
// Prep kernels: fp32 -> bf16 copy, and fp32 [K,N] -> bf16 [N,K] transpose
// ---------------------------------------------------------------------------
__global__ void k_f32_to_bf16(const float* __restrict__ in, bf16* __restrict__ out, int n) {
  int i = blockIdx.x * blockDim.x + threadIdx.x;
  if (i < n) out[i] = (bf16)in[i];
}

__global__ void k_transpose_bf16(const float* __restrict__ in /*[K,N]*/,
                                 bf16* __restrict__ out /*[N,K]*/, int K, int N) {
  int o = blockIdx.x * blockDim.x + threadIdx.x;
  if (o >= K * N) return;
  int n = o / K, kk = o % K;
  out[o] = (bf16)in[(size_t)kk * N + n];
}

// ---------------------------------------------------------------------------
// WMMA GEMM: one wave computes a 16x64 tile (4 accumulators), sharing one
// A-fragment per k-step across 4 WMMAs (4x A-traffic reduction, 4 independent
// WMMAs in flight per k-step for latency hiding).
//   MODE 0: bf16 out, scatter to [B,H,S,dh]    (q, k)
//   MODE 1: bf16 out, scatter to [B,H,dh,S]    (v transposed)
//   MODE 2: f32  out, row-major [M,N]          (align)
// grid = (N/64, M/16), block = 32.
// ---------------------------------------------------------------------------
template <int MODE>
__global__ __launch_bounds__(32) void k_gemm_bf16(const bf16* __restrict__ A,
                                                  const bf16* __restrict__ BT,
                                                  const float* __restrict__ bias,
                                                  void* __restrict__ out,
                                                  int Mn, int Nn, int Kn) {
  int n0 = blockIdx.x * 64;
  int m0 = blockIdx.y * 16;
  const bf16* arow = A + (size_t)m0 * Kn;

  v8f acc[4];
#pragma unroll
  for (int t = 0; t < 4; ++t) acc[t] = v8f{};

  for (int k0 = 0; k0 < Kn; k0 += 32) {
    v16bf a = frag_a_rowmajor(arow, Kn, k0);
#pragma unroll
    for (int t = 0; t < 4; ++t) {
      v16bf b = frag_b_nmajor(BT, Kn, n0 + t * 16, k0);
      acc[t] = WMMA_BF16(a, b, acc[t]);
    }
  }

  int lane  = threadIdx.x & 31;
  int n     = lane & 15;
  int mbase = (lane & 16) ? 8 : 0;
#pragma unroll
  for (int t = 0; t < 4; ++t) {
    int col = n0 + t * 16 + n;
    float bv_ = bias[col];
#pragma unroll
    for (int e = 0; e < 8; ++e) {
      int row = m0 + mbase + e;
      float val = acc[t][e] + bv_;
      if (MODE == 2) {
        ((float*)out)[(size_t)row * Nn + col] = val;
      } else {
        int b_ = row / SS, s = row % SS;
        int h  = col / DHD, dd = col % DHD;
        if (MODE == 0)
          ((bf16*)out)[(((size_t)(b_ * HH + h) * SS + s) * DHD) + dd] = (bf16)val;
        else  // MODE 1: v transposed per head -> [B,H,dh,S]
          ((bf16*)out)[(((size_t)(b_ * HH + h) * DHD + dd) * SS) + s] = (bf16)val;
      }
    }
  }
}

// ---------------------------------------------------------------------------
// cs / gate projections (N = H = 12, too narrow for a 16-wide MMA tile)
// ---------------------------------------------------------------------------
__global__ void k_csgate(const float* __restrict__ hs,
                         const float* __restrict__ Wcs, const float* __restrict__ bcs,
                         const float* __restrict__ Wg,  const float* __restrict__ bg,
                         float* __restrict__ cs, float* __restrict__ gate) {
  int idx = blockIdx.x * blockDim.x + threadIdx.x;
  int total = MM * HH;
  if (idx >= 2 * total) return;
  int which = idx / total;
  int rem   = idx % total;
  int row = rem / HH, h = rem % HH;
  const float* W = which ? Wg : Wcs;
  float acc = which ? bg[h] : bcs[h];
  const float* x = hs + (size_t)row * DD;
  for (int kk = 0; kk < DD; ++kk) acc = fmaf(x[kk], W[(size_t)kk * HH + h], acc);
  if (which) gate[rem] = 1.0f / (1.0f + __expf(-acc));
  else       cs[rem]   = acc;
}

// ---------------------------------------------------------------------------
// Fused attention: per (b, h, 16-query tile), one wave.
//  scores = (q k^T)/sqrt(dh) + cs  -> *gate -> +align -> softmax -> probs out
//  ctx = probs @ v  (v stored transposed [dh,S] per head)
// ---------------------------------------------------------------------------
#define LDP 772  // 768 + 4 pad (distinct banks for the 16 A-fragment rows)

__global__ __launch_bounds__(32) void k_attn(const bf16* __restrict__ q,   // [B,H,S,dh]
                                             const bf16* __restrict__ k,   // [B,H,S,dh]
                                             const bf16* __restrict__ vT,  // [B,H,dh,S]
                                             const float* __restrict__ cs,   // [B*S,H]
                                             const float* __restrict__ gate, // [B*S,H]
                                             const float* __restrict__ align,// [B,S,S]
                                             float* __restrict__ ctx_out,    // [B,S,D]
                                             float* __restrict__ probs_out)  // [B,H,S,S]
{
  __shared__ float sc[16 * LDP];
  __shared__ float csL[16], gateL[16];

  int q0 = blockIdx.x * 16;
  int h  = blockIdx.y;
  int b  = blockIdx.z;
  int lane = threadIdx.x & 31;

  if (lane < 16) {
    csL[lane]   = cs[((size_t)b * SS + q0 + lane) * HH + h];
    gateL[lane] = gate[((size_t)b * SS + q0 + lane) * HH + h];
  }
  __syncthreads();

  const bf16* qbase = q  + ((size_t)(b * HH + h) * SS + q0) * DHD;
  const bf16* kbase = k  + (size_t)(b * HH + h) * SS * DHD;
  const bf16* vbase = vT + (size_t)(b * HH + h) * DHD * SS;

  // q fragments hoisted out of the key loop (K = dh = 64 -> two k-steps)
  v16bf a0 = frag_a_rowmajor(qbase, DHD, 0);
  v16bf a1 = frag_a_rowmajor(qbase, DHD, 32);

  const float scale = 0.125f;  // 1/sqrt(64)
  int n     = lane & 15;
  int mbase = (lane & 16) ? 8 : 0;

  // ---- scores: QK^T over all key tiles, fused bias/gate/align ----
  for (int kt = 0; kt < SS / 16; ++kt) {
    int n0 = kt * 16;
    v16bf b0 = frag_b_nmajor(kbase, DHD, n0, 0);
    v16bf b1 = frag_b_nmajor(kbase, DHD, n0, 32);
    v8f c = {};
    c = WMMA_BF16(a0, b0, c);
    c = WMMA_BF16(a1, b1, c);
#pragma unroll
    for (int e = 0; e < 8; ++e) {
      int m = mbase + e;
      float val = c[e] * scale + csL[m];
      val *= gateL[m];
      val += align[((size_t)b * SS + q0 + m) * SS + n0 + n];
      sc[m * LDP + n0 + n] = val;
    }
  }
  __syncthreads();

  // ---- row softmax (wave32 reductions) + coalesced probs store ----
  for (int r = 0; r < 16; ++r) {
    float mx = -3.402823466e38f;
    for (int j = lane; j < SS; j += 32) mx = fmaxf(mx, sc[r * LDP + j]);
#pragma unroll
    for (int off = 16; off >= 1; off >>= 1) mx = fmaxf(mx, __shfl_xor(mx, off, 32));
    float sum = 0.0f;
    for (int j = lane; j < SS; j += 32) {
      float e = __expf(sc[r * LDP + j] - mx);
      sc[r * LDP + j] = e;
      sum += e;
    }
#pragma unroll
    for (int off = 16; off >= 1; off >>= 1) sum += __shfl_xor(sum, off, 32);
    float inv = 1.0f / sum;
    float* prow = probs_out + ((size_t)(b * HH + h) * SS + q0 + r) * SS;
    for (int j = lane; j < SS; j += 32) {
      float p = sc[r * LDP + j] * inv;
      sc[r * LDP + j] = p;
      prow[j] = p;
    }
  }
  __syncthreads();

  // ---- ctx = probs @ v : A from LDS (bf16-converted), B from vT rows ----
  v8f acc[4];
#pragma unroll
  for (int t = 0; t < 4; ++t) acc[t] = v8f{};
  for (int k0 = 0; k0 < SS; k0 += 32) {
    v16bf a = frag_a_lds_f32(sc, LDP, k0);
#pragma unroll
    for (int t = 0; t < 4; ++t) {
      v16bf bb = frag_b_nmajor(vbase, SS, t * 16, k0);
      acc[t] = WMMA_BF16(a, bb, acc[t]);
    }
  }
#pragma unroll
  for (int t = 0; t < 4; ++t) {
#pragma unroll
    for (int e = 0; e < 8; ++e) {
      int m  = mbase + e;
      int dd = t * 16 + n;
      ctx_out[((size_t)(b * SS + q0 + m)) * DD + h * DHD + dd] = acc[t][e];
    }
  }
}

// ---------------------------------------------------------------------------
// Launcher
// ---------------------------------------------------------------------------
extern "C" void kernel_launch(void* const* d_in, const int* in_sizes, int n_in,
                              void* d_out, int out_size, void* d_ws, size_t ws_size,
                              hipStream_t stream) {
  const float* hs  = (const float*)d_in[0];
  const float* ic  = (const float*)d_in[1];
  const float* Wq  = (const float*)d_in[2];
  const float* bq  = (const float*)d_in[3];
  const float* Wk  = (const float*)d_in[4];
  const float* bk  = (const float*)d_in[5];
  const float* Wv  = (const float*)d_in[6];
  const float* bv  = (const float*)d_in[7];
  const float* Wcs = (const float*)d_in[8];
  const float* bcs = (const float*)d_in[9];
  const float* Wg  = (const float*)d_in[10];
  const float* bg  = (const float*)d_in[11];
  const float* Wa  = (const float*)d_in[12];
  const float* ba  = (const float*)d_in[13];

  // Workspace carve-up (bytes)
  char* ws = (char*)d_ws;
  size_t off = 0;
  auto take = [&](size_t bytes) { char* p = ws + off; off += (bytes + 255) & ~(size_t)255; return p; };
  bf16*  hsb   = (bf16*)take((size_t)MM * DD * sizeof(bf16));   // 9.4 MB
  bf16*  icb   = (bf16*)take((size_t)MM * DD * sizeof(bf16));   // 9.4 MB
  bf16*  wqT   = (bf16*)take((size_t)DD * DD * sizeof(bf16));
  bf16*  wkT   = (bf16*)take((size_t)DD * DD * sizeof(bf16));
  bf16*  wvT   = (bf16*)take((size_t)DD * DD * sizeof(bf16));
  bf16*  waT   = (bf16*)take((size_t)DD * DD * sizeof(bf16));
  bf16*  qb    = (bf16*)take((size_t)MM * DD * sizeof(bf16));   // [B,H,S,dh]
  bf16*  kb    = (bf16*)take((size_t)MM * DD * sizeof(bf16));   // [B,H,S,dh]
  bf16*  vb    = (bf16*)take((size_t)MM * DD * sizeof(bf16));   // [B,H,dh,S]
  float* alg   = (float*)take((size_t)MM * SS * sizeof(float)); // 18.9 MB
  float* csb   = (float*)take((size_t)MM * HH * sizeof(float));
  float* gtb   = (float*)take((size_t)MM * HH * sizeof(float));
  (void)ws_size; (void)n_in; (void)in_sizes; (void)out_size;

  float* ctx_out   = (float*)d_out;
  float* probs_out = ctx_out + (size_t)BB * SS * DD;

  // 1) conversions
  {
    int n = MM * DD;
    k_f32_to_bf16<<<(n + 255) / 256, 256, 0, stream>>>(hs, hsb, n);
    k_f32_to_bf16<<<(n + 255) / 256, 256, 0, stream>>>(ic, icb, n);
    int w = DD * DD;
    k_transpose_bf16<<<(w + 255) / 256, 256, 0, stream>>>(Wq, wqT, DD, DD);
    k_transpose_bf16<<<(w + 255) / 256, 256, 0, stream>>>(Wk, wkT, DD, DD);
    k_transpose_bf16<<<(w + 255) / 256, 256, 0, stream>>>(Wv, wvT, DD, DD);
    k_transpose_bf16<<<(w + 255) / 256, 256, 0, stream>>>(Wa, waT, DD, DD);
  }

  // 2) WMMA projections: q, k, vT, align (16x64 tile per wave)
  {
    dim3 grid(DD / 64, MM / 16), block(32);
    k_gemm_bf16<0><<<grid, block, 0, stream>>>(hsb, wqT, bq, (void*)qb, MM, DD, DD);
    k_gemm_bf16<0><<<grid, block, 0, stream>>>(hsb, wkT, bk, (void*)kb, MM, DD, DD);
    k_gemm_bf16<1><<<grid, block, 0, stream>>>(hsb, wvT, bv, (void*)vb, MM, DD, DD);
    k_gemm_bf16<2><<<grid, block, 0, stream>>>(icb, waT, ba, (void*)alg, MM, DD, DD);
  }

  // 3) cs / gate
  {
    int n = 2 * MM * HH;
    k_csgate<<<(n + 255) / 256, 256, 0, stream>>>(hs, Wcs, bcs, Wg, bg, csb, gtb);
  }

  // 4) fused attention
  {
    dim3 grid(SS / 16, HH, BB), block(32);
    k_attn<<<grid, block, 0, stream>>>(qb, kb, vb, csb, gtb, alg, ctx_out, probs_out);
  }
}
